// LGMixer_20822001451575
// MI455X (gfx1250) — matrix operand order repaired and checked by reading, not compiled, not run on hardware
//
#include <hip/hip_runtime.h>
#include <hip/hip_bf16.h>
#include <math.h>

// MI455X / gfx1250, wave32. All matrix work via v_wmma_f32_16x16x32_bf16.
typedef __attribute__((ext_vector_type(16))) __bf16 v16bf;
typedef __attribute__((ext_vector_type(8)))  float  v8f;

// ---- A-fragment K index map (contiguous: k = 16*khalf + j, matches B fills) ----
__device__ __forceinline__ int a_kmap(int j, int khalf) {
    return 16 * khalf + j;
}

// =====================================================================
// Kernel A: windowed multi-head attention on channels [0,32)
// one block per 8x8 window; 4 waves = 4 heads
// =====================================================================
__global__ __launch_bounds__(128) void lg_attn_kernel(
    const float* __restrict__ xin, const float* __restrict__ w_qkv,
    const float* __restrict__ b_qkv, const float* __restrict__ pos_emb,
    float* __restrict__ cat)
{
    __shared__ __bf16 Xw[64][32];
    __shared__ __bf16 Qs[64][32];
    __shared__ __bf16 Ks[64][32];
    __shared__ __bf16 Vs[64][32];
    __shared__ __bf16 Ps[4][64][64];

    const int t    = threadIdx.x;
    const int wave = t >> 5;
    const int lane = t & 31;
    const int nl   = lane & 15;
    const int hi   = lane >> 4;

    const int win  = blockIdx.x;
    const int b_i  = win >> 10;
    const int hb_i = (win >> 5) & 31;
    const int wb_i = win & 31;
    const int y0 = hb_i * 8, x0 = wb_i * 8;

    // stage window [64 tokens x 32 ch] into LDS as bf16 (float4 loads)
    for (int g = t; g < 512; g += 128) {
        int row = g >> 3, c = (g & 7) << 2;
        int yy = y0 + (row >> 3), xx = x0 + (row & 7);
        const float4 v = *(const float4*)(xin + ((((size_t)b_i * 256 + yy) * 256 + xx) * 64 + c));
        Xw[row][c + 0] = (__bf16)v.x;
        Xw[row][c + 1] = (__bf16)v.y;
        Xw[row][c + 2] = (__bf16)v.z;
        Xw[row][c + 3] = (__bf16)v.w;
    }
    __syncthreads();

    // ---- QKV GEMM: this wave computes rows [16*wave, 16*wave+16), K=32 ----
    const int i0 = wave * 16;
    v16bf Afrag;
#pragma unroll
    for (int j = 0; j < 16; ++j) Afrag[j] = Xw[i0 + nl][a_kmap(j, hi)];

    v8f acc[6];
#pragma unroll
    for (int dt = 0; dt < 6; ++dt) {
        // B[k][d] = w_qkv[d][k]; lane: d = dt*16+nl, k = 16*hi+e (contiguous)
        v16bf Bfrag;
        const float* wr = w_qkv + (dt * 16 + nl) * 32 + 16 * hi;
#pragma unroll
        for (int e = 0; e < 16; ++e) Bfrag[e] = (__bf16)wr[e];
        v8f z;
#pragma unroll
        for (int r = 0; r < 8; ++r) z[r] = 0.0f;
        acc[dt] = __builtin_amdgcn_wmma_f32_16x16x32_bf16(false, Afrag, false, Bfrag,
                                                          (short)0, z, false, false);
    }
    // bias, q-scale (hc^-0.5), scatter to LDS as bf16
#pragma unroll
    for (int dt = 0; dt < 6; ++dt) {
        float bq = b_qkv[dt * 16 + nl];
        float sc = (dt < 2) ? 0.35355339059327373f : 1.0f;
#pragma unroll
        for (int r = 0; r < 8; ++r) {
            float val = (acc[dt][r] + bq) * sc;
            int m = i0 + r + 8 * hi;
            if (dt < 2)       Qs[m][dt * 16 + nl]        = (__bf16)val;
            else if (dt < 4)  Ks[m][(dt - 2) * 16 + nl]  = (__bf16)val;
            else              Vs[m][(dt - 4) * 16 + nl]  = (__bf16)val;
        }
    }
    __syncthreads();

    // ---- per-head attention: wave = head h, channels [8h, 8h+8) ----
    const int h = wave;
    const float* posp = pos_emb + h * 4096;

    // B fragments for q.k^T: B[k][j] = K[j][c=k], K padded 8 -> 32
    v16bf Kb[4];
#pragma unroll
    for (int jt = 0; jt < 4; ++jt) {
        int jcol = jt * 16 + nl;
#pragma unroll
        for (int e = 0; e < 16; ++e) {
            int k = 16 * hi + e;
            Kb[jt][e] = (k < 8) ? Ks[jcol][8 * h + k] : (__bf16)0.0f;
        }
    }

#pragma unroll
    for (int mi = 0; mi < 4; ++mi) {
        v16bf Qa;
#pragma unroll
        for (int j = 0; j < 16; ++j) {
            int k = a_kmap(j, hi);
            Qa[j] = (k < 8) ? Qs[mi * 16 + nl][8 * h + k] : (__bf16)0.0f;
        }
        v8f S[4];
#pragma unroll
        for (int jt = 0; jt < 4; ++jt) {
            v8f z;
#pragma unroll
            for (int r = 0; r < 8; ++r) z[r] = 0.0f;
            S[jt] = __builtin_amdgcn_wmma_f32_16x16x32_bf16(false, Qa, false, Kb[jt],
                                                            (short)0, z, false, false);
        }
        // + pos_emb, then row softmax (row lives in one 16-lane half x 4 tiles)
#pragma unroll
        for (int r = 0; r < 8; ++r) {
            int i = mi * 16 + r + 8 * hi;
            float s0 = S[0][r] + posp[i * 64 +  0 + nl];
            float s1 = S[1][r] + posp[i * 64 + 16 + nl];
            float s2 = S[2][r] + posp[i * 64 + 32 + nl];
            float s3 = S[3][r] + posp[i * 64 + 48 + nl];
            float mx = fmaxf(fmaxf(s0, s1), fmaxf(s2, s3));
#pragma unroll
            for (int d = 1; d < 16; d <<= 1) mx = fmaxf(mx, __shfl_xor(mx, d));
            float e0 = __expf(s0 - mx), e1 = __expf(s1 - mx);
            float e2 = __expf(s2 - mx), e3 = __expf(s3 - mx);
            float sum = e0 + e1 + e2 + e3;
#pragma unroll
            for (int d = 1; d < 16; d <<= 1) sum += __shfl_xor(sum, d);
            float inv = 1.0f / sum;
            Ps[h][i][ 0 + nl] = (__bf16)(e0 * inv);
            Ps[h][i][16 + nl] = (__bf16)(e1 * inv);
            Ps[h][i][32 + nl] = (__bf16)(e2 * inv);
            Ps[h][i][48 + nl] = (__bf16)(e3 * inv);
        }
    }
    __syncthreads();

    // ---- out = P @ V (K=64, 2 k-steps); B cols 8..15 zero-padded ----
    v16bf Vb[2];
#pragma unroll
    for (int ks = 0; ks < 2; ++ks) {
#pragma unroll
        for (int e = 0; e < 16; ++e) {
            int k = ks * 32 + 16 * hi + e;
            Vb[ks][e] = (nl < 8) ? Vs[k][8 * h + nl] : (__bf16)0.0f;
        }
    }
#pragma unroll
    for (int mi = 0; mi < 4; ++mi) {
        v8f O;
#pragma unroll
        for (int r = 0; r < 8; ++r) O[r] = 0.0f;
#pragma unroll
        for (int ks = 0; ks < 2; ++ks) {
            v16bf Pa;
#pragma unroll
            for (int j = 0; j < 16; ++j)
                Pa[j] = Ps[h][mi * 16 + nl][ks * 32 + a_kmap(j, hi)];
            O = __builtin_amdgcn_wmma_f32_16x16x32_bf16(false, Pa, false, Vb[ks],
                                                        (short)0, O, false, false);
        }
        if (nl < 8) {
#pragma unroll
            for (int r = 0; r < 8; ++r) {
                int i = mi * 16 + r + 8 * hi;
                int yy = y0 + (i >> 3), xx = x0 + (i & 7);
                cat[(((size_t)b_i * 256 + yy) * 256 + xx) * 64 + 8 * h + nl] = O[r];
            }
        }
    }
}

// =====================================================================
// FFT helpers: 256-pt radix-2 DIT in LDS, 128 threads, input bit-reversed
// =====================================================================
__device__ __forceinline__ unsigned bitrev8(unsigned x) {
    x = ((x & 0x55u) << 1) | ((x & 0xAAu) >> 1);
    x = ((x & 0x33u) << 2) | ((x & 0xCCu) >> 2);
    x = ((x & 0x0Fu) << 4) | ((x & 0xF0u) >> 4);
    return x;
}

__device__ __forceinline__ void fft256_stages(float* re, float* im, int t, float sign) {
#pragma unroll
    for (int s = 1; s <= 8; ++s) {
        int m = 1 << s, half = m >> 1;
        int g = t >> (s - 1), j = t & (half - 1);
        int i1 = g * m + j, i2 = i1 + half;
        float ang = sign * 6.283185307179586f * (float)j / (float)m;
        float sn, cs;
        __sincosf(ang, &sn, &cs);
        float ur = re[i1], ui = im[i1];
        float vr = re[i2], vi = im[i2];
        float tr = vr * cs - vi * sn;
        float ti = vr * sn + vi * cs;
        re[i1] = ur + tr; im[i1] = ui + ti;
        re[i2] = ur - tr; im[i2] = ui - ti;
        __syncthreads();
    }
}

// Kernel B1: per-(b,ch,y) row rFFT -> bins 0..128
__global__ __launch_bounds__(128) void lg_fft_rows(const float* __restrict__ xin,
                                                   float2* __restrict__ freq)
{
    __shared__ float re[256], im[256];
    const int id = blockIdx.x;
    const int y = id & 255, ch = (id >> 8) & 31, b_i = id >> 13;
    const int t = threadIdx.x;
    for (int i = t; i < 256; i += 128) {
        float v = xin[(((size_t)b_i * 256 + y) * 256 + i) * 64 + 32 + ch];
        int r = bitrev8(i);
        re[r] = v; im[r] = 0.0f;
    }
    __syncthreads();
    fft256_stages(re, im, t, -1.0f);
    size_t base = (((size_t)b_i * 32 + ch) * 256 + y) * 129;
    freq[base + t] = make_float2(re[t], im[t]);
    if (t == 0) freq[base + 128] = make_float2(re[128], im[128]);
}

// Kernel B2: per-(b,ch,kx) column FFT -> amp/phase remap -> inverse column FFT
__global__ __launch_bounds__(128) void lg_fft_cols(float2* __restrict__ freq,
    const float* __restrict__ aw, const float* __restrict__ ab,
    const float* __restrict__ pw, const float* __restrict__ pb)
{
    __shared__ float re[256], im[256];
    const int id = blockIdx.x;
    const int k = id % 129, ch = (id / 129) % 32, b_i = id / (129 * 32);
    const int t = threadIdx.x;
    const size_t base = ((size_t)b_i * 32 + ch) * 256 * 129 + k;
    for (int i = t; i < 256; i += 128) {
        float2 v = freq[base + (size_t)i * 129];
        int r = bitrev8(i);
        re[r] = v.x; im[r] = v.y;
    }
    __syncthreads();
    fft256_stages(re, im, t, -1.0f);
    const float AW = aw[ch], AB = ab[ch], PW = pw[ch], PB = pb[ch];
    for (int i = t; i < 256; i += 128) {
        float x = re[i], y = im[i];
        float amp = sqrtf(x * x + y * y);
        float pha = atan2f(y, x);
        float af = amp * AW + AB;
        float pf = pha * PW + PB;
        float sn, cs;
        __sincosf(pf, &sn, &cs);
        re[i] = af * cs + 2e-8f;   // +1e-8 on real twice in reference
        im[i] = af * sn + 1e-8f;
    }
    __syncthreads();
    for (int i = t; i < 256; i += 128) {   // in-place bit-reverse permute
        int r = bitrev8(i);
        if (i < (int)r) {
            float tr = re[i]; re[i] = re[r]; re[r] = tr;
            float ti = im[i]; im[i] = im[r]; im[r] = ti;
        }
    }
    __syncthreads();
    fft256_stages(re, im, t, 1.0f);
    const float s = 1.0f / 256.0f;
    for (int i = t; i < 256; i += 128)
        freq[base + (size_t)i * 129] = make_float2(re[i] * s, im[i] * s);
}

// Kernel B3: per-(b,ch,y) Hermitian extension + inverse row FFT, |.| -> cat[32:]
__global__ __launch_bounds__(128) void lg_ifft_rows(const float2* __restrict__ freq,
                                                    float* __restrict__ cat)
{
    __shared__ float re[256], im[256];
    const int id = blockIdx.x;
    const int y = id & 255, ch = (id >> 8) & 31, b_i = id >> 13;
    const int t = threadIdx.x;
    const size_t base = (((size_t)b_i * 32 + ch) * 256 + y) * 129;
    {
        float2 v = freq[base + t];
        re[bitrev8(t)] = v.x; im[bitrev8(t)] = v.y;
        float2 w;
        if (t == 0) w = freq[base + 128];
        else { float2 u = freq[base + (128 - t)]; w = make_float2(u.x, -u.y); }
        int r2 = bitrev8(t + 128);
        re[r2] = w.x; im[r2] = w.y;
    }
    __syncthreads();
    fft256_stages(re, im, t, 1.0f);
    const float s = 1.0f / 256.0f;
    for (int i = t; i < 256; i += 128)
        cat[(((size_t)b_i * 256 + y) * 256 + i) * 64 + 32 + ch] = fabsf(re[i] * s);
}

// =====================================================================
// Kernel C: out[m][d] = cat[m][:] . proj_w[d][:] + proj_b[d]  (K=64)
// one wave = 16 rows x 64 cols, bias pre-seeded into the accumulator
// =====================================================================
__global__ __launch_bounds__(256) void lg_proj(const float* __restrict__ cat,
    const float* __restrict__ proj_w, const float* __restrict__ proj_b,
    float* __restrict__ out)
{
    const int t = threadIdx.x;
    const int wave = t >> 5, lane = t & 31, nl = lane & 15, hi = lane >> 4;
    const size_t row0 = (size_t)blockIdx.x * 128 + wave * 16;

    __builtin_prefetch(cat + (row0 + 128) * 64, 0, 1);  // global_prefetch_b8

    v16bf A0, A1;
    const float* arow = cat + (row0 + nl) * 64;
#pragma unroll
    for (int j = 0; j < 16; ++j) {
        int k = a_kmap(j, hi);
        A0[j] = (__bf16)arow[k];
        A1[j] = (__bf16)arow[32 + k];
    }
    v8f acc[4];
#pragma unroll
    for (int dt = 0; dt < 4; ++dt) {
        float bq = proj_b[dt * 16 + nl];
        v8f c;
#pragma unroll
        for (int r = 0; r < 8; ++r) c[r] = bq;
#pragma unroll
        for (int ks = 0; ks < 2; ++ks) {
            v16bf Bf;
            const float* wr = proj_w + (dt * 16 + nl) * 64 + ks * 32 + 16 * hi;
#pragma unroll
            for (int e = 0; e < 16; ++e) Bf[e] = (__bf16)wr[e];
            c = __builtin_amdgcn_wmma_f32_16x16x32_bf16(false, (ks == 0) ? A0 : A1,
                                                        false, Bf, (short)0, c, false, false);
        }
        acc[dt] = c;
    }
#pragma unroll
    for (int dt = 0; dt < 4; ++dt)
#pragma unroll
        for (int r = 0; r < 8; ++r)
            out[(row0 + r + 8 * hi) * 64 + dt * 16 + nl] = acc[dt][r];
}

// =====================================================================
extern "C" void kernel_launch(void* const* d_in, const int* in_sizes, int n_in,
                              void* d_out, int out_size, void* d_ws, size_t ws_size,
                              hipStream_t stream) {
    const float* x       = (const float*)d_in[0];
    const float* w_qkv   = (const float*)d_in[1];
    const float* b_qkv   = (const float*)d_in[2];
    const float* pos_emb = (const float*)d_in[3];
    const float* amp_w   = (const float*)d_in[4];
    const float* amp_b   = (const float*)d_in[5];
    const float* pha_w   = (const float*)d_in[6];
    const float* pha_b   = (const float*)d_in[7];
    const float* proj_w  = (const float*)d_in[8];
    const float* proj_b  = (const float*)d_in[9];
    float*  out  = (float*)d_out;
    float*  cat  = (float*)d_ws;                                   // 128 MiB
    float2* freq = (float2*)((char*)d_ws + (size_t)134217728);     // 67 MiB

    lg_attn_kernel<<<8192, 128, 0, stream>>>(x, w_qkv, b_qkv, pos_emb, cat);
    lg_fft_rows  <<<65536, 128, 0, stream>>>(x, freq);
    lg_fft_cols  <<<33024, 128, 0, stream>>>(freq, amp_w, amp_b, pha_w, pha_b);
    lg_ifft_rows <<<65536, 128, 0, stream>>>(freq, cat);
    lg_proj      <<<4096, 256, 0, stream>>>(cat, proj_w, proj_b, out);
}
